// MultiHeadAttention_17952963297976
// MI455X (gfx1250) — matrix operand order repaired and checked by reading, not compiled, not run on hardware
//
#include <hip/hip_runtime.h>

typedef __attribute__((ext_vector_type(16))) _Float16 v16h;
typedef __attribute__((ext_vector_type(8)))  _Float16 v8h;
typedef __attribute__((ext_vector_type(8)))  float    v8f;
typedef __attribute__((ext_vector_type(4)))  unsigned int u32x4;
typedef __attribute__((ext_vector_type(8)))  int i32x8;
typedef __attribute__((ext_vector_type(4)))  int i32x4;

#define B_    2
#define L_    2048
#define D_    1024
#define H_    16
#define DK_   64
#define M_TOT (B_ * L_)   // 4096

union AFrag { v16h v; v8h h[2]; };

// A fragment via uniform base + 32-bit element offset (saddr+voffset codegen).
// Lane (m = lane&15, hi = lane>>4) holds row m elements [off+hi*16,+16)
// (contiguous k order, matching the B fragments)
static __device__ __forceinline__ v16h load_a_off(const _Float16* base,
                                                  unsigned off, int hi) {
    AFrag f;
    f.h[0] = *(const v8h*)(base + off + hi * 16);
    f.h[1] = *(const v8h*)(base + off + hi * 16 + 8);
    return f.v;
}

// B fragment in "Bt" form (Bt[n][k] row-major, computes A @ Bt^T):
// lane (n = lane&15, hi) holds Bt row n halves [off+hi*16, +16): one 32B chunk
static __device__ __forceinline__ v16h load_bt_off(const _Float16* base,
                                                   unsigned off, int hi) {
    return *(const v16h*)(base + off + hi * 16);
}

static __device__ __forceinline__ v8f wmma_f16(v16h a, v16h b, v8f c) {
    return __builtin_amdgcn_wmma_f32_16x16x32_f16(
        false, a, false, b, (short)0, c, false, false);
}

// ---- Tensor Data Mover: 2D tile (16-bit elements) global -> LDS -------------
// Descriptor per CDNA5 ISA (group0 128b, group1 256b).
// clang-23 toolchain: 6-arg builtin (g0, g1, g2, g3, extra, cpol).
static __device__ __forceinline__ void tdm_load_2d(unsigned lds_off_bytes,
                                                   const void* gaddr,
                                                   unsigned tile_w,   // elems/row
                                                   unsigned tile_h,   // rows
                                                   unsigned row_stride_elems) {
    unsigned long long ga = (unsigned long long)(uintptr_t)gaddr;
    u32x4 g0;
    g0[0] = 1u;                                   // count=1, user mode, no gather
    g0[1] = lds_off_bytes;                        // lds_addr [63:32]
    g0[2] = (unsigned)(ga & 0xffffffffu);         // global_addr [95:64]
    g0[3] = (unsigned)((ga >> 32) & 0x1ffffffu)   // global_addr [120:96]
          | (2u << 30);                           // type=2 ("image") [127:126]
    i32x8 g1;
    g1[0] = (int)(1u << 16);                      // wg_mask=0 | data_size=1 (2B)
    g1[1] = (int)((tile_w & 0xffffu) << 16);      // tensor_dim0[15:0] @ [63:48]
    g1[2] = (int)(((tile_w >> 16) & 0xffffu) |    // tensor_dim0[31:16]
                  ((tile_h & 0xffffu) << 16));    // tensor_dim1[15:0]
    g1[3] = (int)(((tile_h >> 16) & 0xffffu) |    // tensor_dim1[31:16]
                  ((tile_w & 0xffffu) << 16));    // tile_dim0
    g1[4] = (int)(tile_h & 0xffffu);              // tile_dim1 | tile_dim2=0
    g1[5] = (int)row_stride_elems;                // tensor_dim0_stride[31:0]
    g1[6] = 0;                                    // stride[47:32] | dim1_stride lo
    g1[7] = 0;                                    // dim1_stride hi (2D: unused)
    i32x4 z4 = {0, 0, 0, 0};
    i32x8 z8 = {0, 0, 0, 0, 0, 0, 0, 0};
    __builtin_amdgcn_tensor_load_to_lds(g0, g1, z4, z4, z8, 0);
}

// ---------------------------------------------------------------- convert
__global__ void cvt_f32_f16(const float* __restrict__ in,
                            _Float16* __restrict__ out, int n) {
    int i = blockIdx.x * blockDim.x + threadIdx.x;
    int stride = gridDim.x * blockDim.x;
    for (; i < n; i += stride) out[i] = (_Float16)in[i];
}

// ---------------------------------------------------------------- generic GEMM
// Each wave computes a 32(M) x 64(N) tile of A(MxK,f16) @ Bt(NxK,f16)^T.
// mode 0: Q -> out16[b,h,l,dk] (scaled 1/8)   mode 1: K -> out16[b,h,l,dk]
// mode 2: V -> out16[b,h,dk,l] (transposed)   mode 3: f32 row-major out32
__global__ void gemm_wmma_kernel(const _Float16* __restrict__ A,
                                 const _Float16* __restrict__ Bt,
                                 _Float16* __restrict__ out16,
                                 float* __restrict__ out32,
                                 int M, int N, int K, int mode, float scale) {
    const int lane = threadIdx.x & 31;
    const int hi   = lane >> 4;
    const int ln   = lane & 15;
    const int wid  = (blockIdx.x * blockDim.x + threadIdx.x) >> 5;
    const int ntN  = N >> 6;
    const int tm   = wid / ntN;
    const int tn   = wid - tm * ntN;
    if (tm * 32 >= M) return;

    // 32-bit element offsets; K-step stays in the scalar unit
    const unsigned ar0 = (unsigned)(tm * 32 + ln) * (unsigned)K;
    const unsigned ar1 = ar0 + (unsigned)(16 * K);
    unsigned br[4];
#pragma unroll
    for (int j = 0; j < 4; ++j)
        br[j] = (unsigned)(tn * 64 + j * 16 + ln) * (unsigned)K;

    v8f acc[2][4] = {};

#pragma unroll 1
    for (int k0 = 0; k0 < K; k0 += 32) {
        // issue all loads for this K-step first, then the 8 WMMAs
        v16h a0 = load_a_off(A, ar0 + (unsigned)k0, hi);
        v16h a1 = load_a_off(A, ar1 + (unsigned)k0, hi);
        v16h b[4];
#pragma unroll
        for (int j = 0; j < 4; ++j)
            b[j] = load_bt_off(Bt, br[j] + (unsigned)k0, hi);
        if (k0 + 256 < K) __builtin_prefetch(A + ar0 + k0 + 256, 0, 0);
#pragma unroll
        for (int j = 0; j < 4; ++j) acc[0][j] = wmma_f16(a0, b[j], acc[0][j]);
#pragma unroll
        for (int j = 0; j < 4; ++j) acc[1][j] = wmma_f16(a1, b[j], acc[1][j]);
    }

#pragma unroll
    for (int s = 0; s < 2; ++s)
#pragma unroll
        for (int j = 0; j < 4; ++j)
#pragma unroll
            for (int r = 0; r < 8; ++r) {
                int m = tm * 32 + s * 16 + r + 8 * hi;
                int n = tn * 64 + j * 16 + ln;
                float v = acc[s][j][r] * scale;
                if (mode == 3) {
                    __builtin_nontemporal_store(v, &out32[(size_t)m * N + n]);
                } else {
                    int bb = m >> 11, l = m & (L_ - 1);
                    int hh = n >> 6,  dk = n & 63;
                    size_t idx;
                    if (mode == 2) idx = (((size_t)(bb * H_ + hh)) * DK_ + dk) * L_ + l;
                    else           idx = (((size_t)(bb * H_ + hh)) * L_ + l) * DK_ + dk;
                    out16[idx] = (_Float16)v;
                }
            }
}

// ---------------------------------------------------------------- flash attention
// 8 waves / block; each wave owns a 16-row Q tile of one (b,h); the block
// shares TDM-staged, double-buffered K/V tiles (32 keys) in LDS.
__global__ void flash_attn_kernel(const _Float16* __restrict__ Qh,
                                  const _Float16* __restrict__ Kh,
                                  const _Float16* __restrict__ Vt,
                                  const float* __restrict__ bias,
                                  const float* __restrict__ mask,
                                  _Float16* __restrict__ ctx16) {
    __shared__ __attribute__((aligned(32))) _Float16 kbuf[2][32 * 64]; // [key][d]
    __shared__ __attribute__((aligned(32))) _Float16 vbuf[2][64 * 32]; // [d][key]
    __shared__ __attribute__((aligned(32))) _Float16 pbuf[8][16 * 32];

    const int lane = threadIdx.x & 31;
    const int hi   = lane >> 4;
    const int ln   = lane & 15;
    const int widx = threadIdx.x >> 5;
    const int bh   = blockIdx.x >> 4;               // 16 blocks per (b,h)
    const int qt   = ((blockIdx.x & 15) << 3) + widx;
    const int b    = bh >> 4;
    const int h    = bh & 15;
    const int q0   = qt * 16;

    const unsigned qoff = (unsigned)((bh * L_ + q0 + ln) * DK_);
    v16h aq0 = load_a_off(Qh, qoff, hi);
    v16h aq1 = load_a_off(Qh, qoff + 32u, hi);

    v8f o[4] = {};
    float rmax[8], rsum[8];
#pragma unroll
    for (int r = 0; r < 8; ++r) { rmax[r] = -1e30f; rsum[r] = 0.0f; }

    _Float16* pw = pbuf[widx];
    const float* biasq = bias + ((size_t)bh * L_ + q0) * L_;
    const _Float16* kg = Kh + (size_t)bh * L_ * DK_;
    const _Float16* vg = Vt + (size_t)bh * DK_ * L_;

    // prologue: wave 0 issues TDM for key block 0 into buffer 0
    if (widx == 0) {
        tdm_load_2d((unsigned)(uintptr_t)&kbuf[0][0], kg, DK_, 32, DK_);
        tdm_load_2d((unsigned)(uintptr_t)&vbuf[0][0], vg, 32, DK_, L_);
    }

    int cur = 0;
    for (int k0 = 0; k0 < L_; k0 += 32) {
        if (widx == 0) __builtin_amdgcn_s_wait_tensorcnt(0);
        __syncthreads();  // buf[cur] ready for all; prior reads of buf[cur^1] done
        if (widx == 0 && k0 + 32 < L_) {
            tdm_load_2d((unsigned)(uintptr_t)&kbuf[cur ^ 1][0],
                        kg + (size_t)(k0 + 32) * DK_, DK_, 32, DK_);
            tdm_load_2d((unsigned)(uintptr_t)&vbuf[cur ^ 1][0],
                        vg + (k0 + 32), 32, DK_, L_);
        }

        // ---- S = (Q/s) @ K^T for keys [k0, k0+32), K tile from LDS
        v8f s[2];
#pragma unroll
        for (int j = 0; j < 2; ++j) {
            const _Float16* krow = &kbuf[cur][(j * 16 + ln) * DK_];
            v16h bk0 = *(const v16h*)(krow + hi * 16);
            v16h bk1 = *(const v16h*)(krow + 32 + hi * 16);
            v8f t = {};
            t = wmma_f16(aq0, bk0, t);
            t = wmma_f16(aq1, bk1, t);
            s[j] = t;
        }
        // ---- + bias + mask: bias streams once -> nontemporal loads (keep L2
        //      for K/V/weights), coalesced in C-fragment layout
#pragma unroll
        for (int j = 0; j < 2; ++j) {
            float mv = mask[b * L_ + k0 + j * 16 + ln];
            unsigned boff = (unsigned)(hi * 8 * L_ + k0 + j * 16 + ln);
#pragma unroll
            for (int r = 0; r < 8; ++r)
                s[j][r] += __builtin_nontemporal_load(biasq + boff + (unsigned)(r * L_)) + mv;
        }
        // ---- online softmax (row stats per (r,hi); reduce over 16 lanes)
        float alpha[8];
#pragma unroll
        for (int r = 0; r < 8; ++r) {
            float tv = fmaxf(s[0][r], s[1][r]);
            tv = fmaxf(tv, __shfl_xor(tv, 1, 32));
            tv = fmaxf(tv, __shfl_xor(tv, 2, 32));
            tv = fmaxf(tv, __shfl_xor(tv, 4, 32));
            tv = fmaxf(tv, __shfl_xor(tv, 8, 32));
            float nm = fmaxf(rmax[r], tv);
            alpha[r] = __expf(rmax[r] - nm);
            rmax[r]  = nm;
        }
#pragma unroll
        for (int j = 0; j < 2; ++j)
#pragma unroll
            for (int r = 0; r < 8; ++r)
                s[j][r] = __expf(s[j][r] - rmax[r]);
#pragma unroll
        for (int r = 0; r < 8; ++r) {
            float ps = s[0][r] + s[1][r];
            ps += __shfl_xor(ps, 1, 32);
            ps += __shfl_xor(ps, 2, 32);
            ps += __shfl_xor(ps, 4, 32);
            ps += __shfl_xor(ps, 8, 32);
            rsum[r] = rsum[r] * alpha[r] + ps;
        }
#pragma unroll
        for (int tt = 0; tt < 4; ++tt)
#pragma unroll
            for (int r = 0; r < 8; ++r)
                o[tt][r] *= alpha[r];
        // ---- P: C-layout -> per-wave LDS -> A fragment (LDS in-order per wave)
#pragma unroll
        for (int j = 0; j < 2; ++j)
#pragma unroll
            for (int r = 0; r < 8; ++r)
                pw[(r + 8 * hi) * 32 + j * 16 + ln] = (_Float16)s[j][r];
        {
            const _Float16* prow = pw + ln * 32;
            AFrag pf;
            pf.h[0] = *(const v8h*)(prow + hi * 16);
            pf.h[1] = *(const v8h*)(prow + hi * 16 + 8);
            // ---- O += P @ V, V tile from LDS (rows = d, 32 keys)
#pragma unroll
            for (int tt = 0; tt < 4; ++tt) {
                const _Float16* vrow = &vbuf[cur][(tt * 16 + ln) * 32];
                v16h bv = *(const v16h*)(vrow + hi * 16);
                o[tt] = wmma_f16(pf.v, bv, o[tt]);
            }
        }
        cur ^= 1;
    }

    // ---- normalize and store ctx as [B, L, H*DK] f16
#pragma unroll
    for (int tt = 0; tt < 4; ++tt)
#pragma unroll
        for (int r = 0; r < 8; ++r) {
            float inv = 1.0f / rsum[r];
            int row = b * L_ + q0 + r + 8 * hi;
            int col = h * DK_ + tt * 16 + ln;
            ctx16[(size_t)row * (H_ * DK_) + col] = (_Float16)(o[tt][r] * inv);
        }
}

// ---------------------------------------------------------------- launch
extern "C" void kernel_launch(void* const* d_in, const int* in_sizes, int n_in,
                              void* d_out, int out_size, void* d_ws, size_t ws_size,
                              hipStream_t stream) {
    (void)in_sizes; (void)n_in; (void)out_size; (void)ws_size;
    const float* x    = (const float*)d_in[0];
    const float* Wq   = (const float*)d_in[1];
    const float* Wk   = (const float*)d_in[2];
    const float* Wv   = (const float*)d_in[3];
    const float* Wo   = (const float*)d_in[4];
    const float* bias = (const float*)d_in[5];
    const float* mask = (const float*)d_in[6];
    float* out = (float*)d_out;

    char* ws = (char*)d_ws;
    size_t off = 0;
    auto alloc = [&](size_t bytes) {
        void* p = ws + off;
        off += (bytes + 255) & ~(size_t)255;
        return p;
    };
    _Float16* x16  = (_Float16*)alloc((size_t)M_TOT * D_ * 2);
    _Float16* wq16 = (_Float16*)alloc((size_t)D_ * D_ * 2);
    _Float16* wk16 = (_Float16*)alloc((size_t)D_ * D_ * 2);
    _Float16* wv16 = (_Float16*)alloc((size_t)D_ * D_ * 2);
    _Float16* wo16 = (_Float16*)alloc((size_t)D_ * D_ * 2);
    _Float16* qh   = (_Float16*)alloc((size_t)B_ * H_ * L_ * DK_ * 2);
    _Float16* kh   = (_Float16*)alloc((size_t)B_ * H_ * L_ * DK_ * 2);
    _Float16* vt   = (_Float16*)alloc((size_t)B_ * H_ * L_ * DK_ * 2);
    _Float16* ctx  = (_Float16*)alloc((size_t)M_TOT * D_ * 2);

    cvt_f32_f16<<<1024, 256, 0, stream>>>(x,  x16,  M_TOT * D_);
    cvt_f32_f16<<<512,  256, 0, stream>>>(Wq, wq16, D_ * D_);
    cvt_f32_f16<<<512,  256, 0, stream>>>(Wk, wk16, D_ * D_);
    cvt_f32_f16<<<512,  256, 0, stream>>>(Wv, wv16, D_ * D_);
    cvt_f32_f16<<<512,  256, 0, stream>>>(Wo, wo16, D_ * D_);

    const int gemmBlocks = (M_TOT / 32) * (D_ / 64) / 4;  // 512 blocks x 4 waves
    gemm_wmma_kernel<<<gemmBlocks, 128, 0, stream>>>(x16, wq16, qh, nullptr,
                                                     M_TOT, D_, D_, 0, 0.125f);
    gemm_wmma_kernel<<<gemmBlocks, 128, 0, stream>>>(x16, wk16, kh, nullptr,
                                                     M_TOT, D_, D_, 1, 1.0f);
    gemm_wmma_kernel<<<gemmBlocks, 128, 0, stream>>>(x16, wv16, vt, nullptr,
                                                     M_TOT, D_, D_, 2, 1.0f);

    const int flashBlocks = (B_ * H_ * (L_ / 16)) / 8;    // 512 blocks x 8 waves
    flash_attn_kernel<<<flashBlocks, 256, 0, stream>>>(qh, kh, vt, bias, mask, ctx);

    gemm_wmma_kernel<<<gemmBlocks, 128, 0, stream>>>(ctx, wo16, nullptr, out,
                                                     M_TOT, D_, D_, 3, 1.0f);
}
